// ParticleAttention_14757507629938
// MI455X (gfx1250) — matrix operand order repaired and checked by reading, not compiled, not run on hardware
//
#include <hip/hip_runtime.h>
#include <hip/hip_bf16.h>

typedef __attribute__((ext_vector_type(16))) __bf16 v16bf;
typedef __attribute__((ext_vector_type(8)))  float  v8f;

#define B_   8
#define T_   64
#define C_   512
#define H_   8
#define DH_  64
#define NQ_  1024   /* N*T  */
#define NK_  2048   /* (N+NC)*T */
#define LPAD 40     /* LDS row stride (bf16 elems) for P tiles */

// ---------------------------------------------------------------------------
// WMMA helper: D = A*B + C  (bf16 in, f32 accumulate)
// ---------------------------------------------------------------------------
static __device__ __forceinline__ v8f wmma_bf16(v16bf a, v16bf b, v8f c) {
  return __builtin_amdgcn_wmma_f32_16x16x32_bf16(false, a, false, b,
                                                 (short)0, c, false, false);
}

// A-operand tile: 16(M) x 32(K), memory row-major [M][K], ld in elements.
// Lane L: row = L%16, half = L/16. V0-3 <- K=8*half..8*half+7, V4-7 <- +16.
static __device__ __forceinline__ v16bf load_tileA(const __hip_bfloat16* base,
                                                   int ld, int lane) {
  int r = lane & 15, hf = lane >> 4;
  const unsigned char* p = (const unsigned char*)(base + (size_t)r * ld + 8 * hf);
  union { uint4 u[2]; v16bf v; } t;
  t.u[0] = *(const uint4*)(p);
  t.u[1] = *(const uint4*)(p + 32);
  return t.v;
}

// B-operand tile: 32(K) x 16(N), loaded from B^T stored row-major [N][K].
// Lane L: n = L%16, half = L/16. V0-3 <- K=8*half..8*half+7, V4-7 <- +16.
static __device__ __forceinline__ v16bf load_tileB(const __hip_bfloat16* base,
                                                   int ld, int lane) {
  int r = lane & 15, hf = lane >> 4;
  const unsigned char* p = (const unsigned char*)(base + (size_t)r * ld + 8 * hf);
  union { uint4 u[2]; v16bf v; } t;
  t.u[0] = *(const uint4*)(p);
  t.u[1] = *(const uint4*)(p + 32);
  return t.v;
}

static __device__ __forceinline__ void split_store(float v, __hip_bfloat16* hi,
                                                   __hip_bfloat16* lo, size_t i) {
  __hip_bfloat16 h = __float2bfloat16(v);
  hi[i] = h;
  lo[i] = __float2bfloat16(v - __bfloat162float(h));
}

// ---------------------------------------------------------------------------
// fp32 -> bf16 hi/lo split kernels
// ---------------------------------------------------------------------------
__global__ void split_kv_kernel(const float* __restrict__ x,
                                const float* __restrict__ c,
                                __hip_bfloat16* __restrict__ hi,
                                __hip_bfloat16* __restrict__ lo) {
  size_t g = (size_t)blockIdx.x * blockDim.x + threadIdx.x;       // over B*32*T*C
  const size_t per = (size_t)T_ * C_;                              // 32768
  size_t b    = g / (32 * per);
  size_t rest = g % (32 * per);
  size_t n    = rest / per;
  size_t w    = rest % per;
  float v = (n < 16) ? x[(b * 16 + n) * per + w]
                     : c[(b * 16 + (n - 16)) * per + w];
  split_store(v, hi, lo, g);
}

__global__ void split_arr_kernel(const float* __restrict__ a,
                                 __hip_bfloat16* __restrict__ hi,
                                 __hip_bfloat16* __restrict__ lo, int n) {
  int g = blockIdx.x * blockDim.x + threadIdx.x;
  if (g < n) split_store(a[g], hi, lo, (size_t)g);
}

// ---------------------------------------------------------------------------
// Generic batched GEMM:  C[b][m][n] = sum_k A[b][m][k] * W[n][k]
// A, W in bf16 hi/lo (bf16x3 products, f32 accumulate).
// One wave computes a 16x64 output slab; block = 4 waves => 64x64 per block.
// mode 0: store bf16 hi/lo row-major        (Q, K)
// mode 1: store bf16 hi/lo transposed (n,m) (V^T)
// mode 2: store f32 row-major               (final output)
// ---------------------------------------------------------------------------
__global__ __launch_bounds__(128) void gemm_bf16x3_kernel(
    const __hip_bfloat16* __restrict__ Ahi, const __hip_bfloat16* __restrict__ Alo,
    long aBatch, int lda,
    const __hip_bfloat16* __restrict__ Whi, const __hip_bfloat16* __restrict__ Wlo,
    int ldw,
    __hip_bfloat16* __restrict__ Chi, __hip_bfloat16* __restrict__ Clo,
    float* __restrict__ Cf,
    long cBatch, int ldc, int K, int mode) {
  int lane  = threadIdx.x & 31;
  int w     = threadIdx.x >> 5;
  int b     = blockIdx.z;
  int mTile = blockIdx.y * 64 + w * 16;
  int nBase = blockIdx.x * 64;

  const __hip_bfloat16* aH = Ahi + (size_t)b * aBatch + (size_t)mTile * lda;
  const __hip_bfloat16* aL = Alo + (size_t)b * aBatch + (size_t)mTile * lda;

  v8f acc[4] = {v8f{}, v8f{}, v8f{}, v8f{}};

  for (int k = 0; k < K; k += 32) {
    v16bf ah = load_tileA(aH + k, lda, lane);
    v16bf al = load_tileA(aL + k, lda, lane);
#pragma unroll
    for (int j = 0; j < 4; ++j) {
      const size_t woff = (size_t)(nBase + j * 16) * ldw + k;
      v16bf bh = load_tileB(Whi + woff, ldw, lane);
      v16bf bl = load_tileB(Wlo + woff, ldw, lane);
      v8f cacc = acc[j];
      cacc = wmma_bf16(ah, bh, cacc);
      cacc = wmma_bf16(ah, bl, cacc);
      cacc = wmma_bf16(al, bh, cacc);
      acc[j] = cacc;
    }
  }

  int r2 = lane & 15, hf = lane >> 4;
#pragma unroll
  for (int j = 0; j < 4; ++j) {
#pragma unroll
    for (int r = 0; r < 8; ++r) {
      float v = acc[j][r];
      int m = mTile + r + 8 * hf;
      int n = nBase + j * 16 + r2;
      if (mode == 2) {
        Cf[(size_t)b * cBatch + (size_t)m * ldc + n] = v;
      } else {
        size_t idx = (mode == 0)
                         ? ((size_t)b * cBatch + (size_t)m * ldc + n)
                         : ((size_t)b * cBatch + (size_t)n * ldc + m);
        split_store(v, Chi, Clo, idx);
      }
    }
  }
}

// ---------------------------------------------------------------------------
// Flash-style attention. Block = (b, 16-query tile), 8 waves = 8 heads.
// Wave: S = (Q Kt)/8 (bf16x3), mask, online softmax, P via LDS -> A-layout,
// Y += P V (bf16x3).  Writes Y as bf16 hi/lo [B*NQ, C] for final projection.
// ---------------------------------------------------------------------------
__global__ __launch_bounds__(256) void attn_kernel(
    const __hip_bfloat16* __restrict__ Qhi, const __hip_bfloat16* __restrict__ Qlo,
    const __hip_bfloat16* __restrict__ Khi, const __hip_bfloat16* __restrict__ Klo,
    const __hip_bfloat16* __restrict__ Vthi, const __hip_bfloat16* __restrict__ Vtlo,
    const unsigned char* __restrict__ mask,
    __hip_bfloat16* __restrict__ Yhi, __hip_bfloat16* __restrict__ Ylo) {
  __shared__ __hip_bfloat16 sm[H_][2][16][LPAD];   // per-wave P tiles (hi, lo)

  int lane = threadIdx.x & 31;
  int h    = threadIdx.x >> 5;
  int b    = blockIdx.x;
  int q0   = blockIdx.y * 16;
  int r2 = lane & 15, hf = lane >> 4;

  const __hip_bfloat16* qbH = Qhi + ((size_t)b * NQ_ + q0) * C_ + h * DH_;
  const __hip_bfloat16* qbL = Qlo + ((size_t)b * NQ_ + q0) * C_ + h * DH_;
  v16bf qh0 = load_tileA(qbH,      C_, lane);
  v16bf qh1 = load_tileA(qbH + 32, C_, lane);
  v16bf ql0 = load_tileA(qbL,      C_, lane);
  v16bf ql1 = load_tileA(qbL + 32, C_, lane);

  v8f acc[4] = {v8f{}, v8f{}, v8f{}, v8f{}};
  float mrow[8], lrow[8];
#pragma unroll
  for (int r = 0; r < 8; ++r) { mrow[r] = -__builtin_inff(); lrow[r] = 0.0f; }

  for (int kb = 0; kb < NK_; kb += 32) {
    const __hip_bfloat16* kH = Khi + ((size_t)b * NK_ + kb) * C_ + h * DH_;
    const __hip_bfloat16* kL = Klo + ((size_t)b * NK_ + kb) * C_ + h * DH_;
    if (kb + 32 < NK_) {
      __builtin_prefetch(kH + 32 * C_, 0, 1);
      __builtin_prefetch(kL + 32 * C_, 0, 1);
    }

    // ---- scores: s0 = keys kb..kb+15, s1 = keys kb+16..kb+31 --------------
    v8f s0 = v8f{}, s1 = v8f{};
    {
      v16bf bh = load_tileB(kH, C_, lane);
      v16bf bl = load_tileB(kL, C_, lane);
      s0 = wmma_bf16(qh0, bh, s0); s0 = wmma_bf16(qh0, bl, s0); s0 = wmma_bf16(ql0, bh, s0);
      bh = load_tileB(kH + 32, C_, lane);
      bl = load_tileB(kL + 32, C_, lane);
      s0 = wmma_bf16(qh1, bh, s0); s0 = wmma_bf16(qh1, bl, s0); s0 = wmma_bf16(ql1, bh, s0);
      bh = load_tileB(kH + 16 * C_, C_, lane);
      bl = load_tileB(kL + 16 * C_, C_, lane);
      s1 = wmma_bf16(qh0, bh, s1); s1 = wmma_bf16(qh0, bl, s1); s1 = wmma_bf16(ql0, bh, s1);
      bh = load_tileB(kH + 16 * C_ + 32, C_, lane);
      bl = load_tileB(kL + 16 * C_ + 32, C_, lane);
      s1 = wmma_bf16(qh1, bh, s1); s1 = wmma_bf16(qh1, bl, s1); s1 = wmma_bf16(ql1, bh, s1);
    }

    // ---- mask + online softmax -------------------------------------------
#pragma unroll
    for (int r = 0; r < 8; ++r) {
      int m = r + 8 * hf;
      const unsigned char* mp =
          mask + ((size_t)b * NQ_ + q0 + m) * (size_t)NK_ + kb + r2;
      float v0 = s0[r] * 0.125f;            // 1/sqrt(64)
      float v1 = s1[r] * 0.125f;
      if (mp[0])  v0 = -1e30f;
      if (mp[16]) v1 = -1e30f;

      float t = fmaxf(v0, v1);              // row max across 16 lanes
      t = fmaxf(t, __shfl_xor(t, 1, 32));
      t = fmaxf(t, __shfl_xor(t, 2, 32));
      t = fmaxf(t, __shfl_xor(t, 4, 32));
      t = fmaxf(t, __shfl_xor(t, 8, 32));
      float newm = fmaxf(mrow[r], t);
      float sc = __expf(mrow[r] - newm);
      mrow[r] = newm;

      float p0 = __expf(v0 - newm);
      float p1 = __expf(v1 - newm);
      float ss = p0 + p1;                   // row sum across 16 lanes
      ss += __shfl_xor(ss, 1, 32);
      ss += __shfl_xor(ss, 2, 32);
      ss += __shfl_xor(ss, 4, 32);
      ss += __shfl_xor(ss, 8, 32);
      lrow[r] = lrow[r] * sc + ss;
#pragma unroll
      for (int d = 0; d < 4; ++d) acc[d][r] *= sc;

      // P (hi/lo bf16) -> LDS as 16x32 row-major tile
      __hip_bfloat16 h0 = __float2bfloat16(p0);
      sm[h][0][m][r2]      = h0;
      sm[h][1][m][r2]      = __float2bfloat16(p0 - __bfloat162float(h0));
      __hip_bfloat16 h1 = __float2bfloat16(p1);
      sm[h][0][m][16 + r2] = h1;
      sm[h][1][m][16 + r2] = __float2bfloat16(p1 - __bfloat162float(h1));
    }
    asm volatile("" ::: "memory");   // keep P stores before A-layout reads (DS is in-order per wave)

    // ---- Y += P * V (V^T rows are B^T layout: ld = NK_) -------------------
    v16bf phi = load_tileA(&sm[h][0][0][0], LPAD, lane);
    v16bf plo = load_tileA(&sm[h][1][0][0], LPAD, lane);
#pragma unroll
    for (int d = 0; d < 4; ++d) {
      const size_t voff = ((size_t)b * C_ + h * DH_ + d * 16) * NK_ + kb;
      v16bf vh = load_tileB(Vthi + voff, NK_, lane);
      v16bf vl = load_tileB(Vtlo + voff, NK_, lane);
      v8f cacc = acc[d];
      cacc = wmma_bf16(phi, vh, cacc);
      cacc = wmma_bf16(phi, vl, cacc);
      cacc = wmma_bf16(plo, vh, cacc);
      acc[d] = cacc;
    }
  }

  // ---- normalize and store Y (bf16 hi/lo, row-major [B*NQ, C]) ------------
#pragma unroll
  for (int d = 0; d < 4; ++d) {
#pragma unroll
    for (int r = 0; r < 8; ++r) {
      float v = acc[d][r] / lrow[r];
      size_t idx = ((size_t)b * NQ_ + q0 + r + 8 * hf) * C_ + h * DH_ + d * 16 + r2;
      split_store(v, Yhi, Ylo, idx);
    }
  }
}

// ---------------------------------------------------------------------------
// Host-side launch
// ---------------------------------------------------------------------------
extern "C" void kernel_launch(void* const* d_in, const int* in_sizes, int n_in,
                              void* d_out, int out_size, void* d_ws, size_t ws_size,
                              hipStream_t stream) {
  const float* x  = (const float*)d_in[0];
  const float* c  = (const float*)d_in[1];
  const unsigned char* mask = (const unsigned char*)d_in[2];
  const float* Wq = (const float*)d_in[3];
  const float* Wk = (const float*)d_in[4];
  const float* Wv = (const float*)d_in[5];
  const float* Wp = (const float*)d_in[6];
  float* out = (float*)d_out;

  char* ws = (char*)d_ws;
  size_t off = 0;
  auto alloc_bf16 = [&](size_t nel) -> __hip_bfloat16* {
    __hip_bfloat16* p = (__hip_bfloat16*)(ws + off);
    off = (off + nel * sizeof(__hip_bfloat16) + 255) & ~(size_t)255;
    return p;
  };

  const size_t kvN = (size_t)B_ * 32 * T_ * C_;   // 8,388,608
  const size_t wN  = (size_t)C_ * C_;             // 262,144
  const size_t qN  = (size_t)B_ * NQ_ * C_;       // 4,194,304
  const size_t kN  = (size_t)B_ * NK_ * C_;       // 8,388,608

  __hip_bfloat16 *kvh = alloc_bf16(kvN), *kvl = alloc_bf16(kvN);
  __hip_bfloat16 *wqh = alloc_bf16(wN),  *wql = alloc_bf16(wN);
  __hip_bfloat16 *wkh = alloc_bf16(wN),  *wkl = alloc_bf16(wN);
  __hip_bfloat16 *wvh = alloc_bf16(wN),  *wvl = alloc_bf16(wN);
  __hip_bfloat16 *wph = alloc_bf16(wN),  *wpl = alloc_bf16(wN);
  __hip_bfloat16 *qh  = alloc_bf16(qN),  *ql  = alloc_bf16(qN);
  __hip_bfloat16 *kh  = alloc_bf16(kN),  *kl  = alloc_bf16(kN);
  __hip_bfloat16 *vth = alloc_bf16(kN),  *vtl = alloc_bf16(kN);
  __hip_bfloat16 *yh  = alloc_bf16(qN),  *yl  = alloc_bf16(qN);

  // 1) fp32 -> bf16 hi/lo splits
  split_kv_kernel<<<dim3((unsigned)(kvN / 256)), 256, 0, stream>>>(x, c, kvh, kvl);
  split_arr_kernel<<<dim3((unsigned)((wN + 255) / 256)), 256, 0, stream>>>(Wq, wqh, wql, (int)wN);
  split_arr_kernel<<<dim3((unsigned)((wN + 255) / 256)), 256, 0, stream>>>(Wk, wkh, wkl, (int)wN);
  split_arr_kernel<<<dim3((unsigned)((wN + 255) / 256)), 256, 0, stream>>>(Wv, wvh, wvl, (int)wN);
  split_arr_kernel<<<dim3((unsigned)((wN + 255) / 256)), 256, 0, stream>>>(Wp, wph, wpl, (int)wN);

  // 2) projections
  // Q = x @ Wq^T   (A = first 1024 rows of each kv batch)
  gemm_bf16x3_kernel<<<dim3(C_ / 64, NQ_ / 64, B_), 128, 0, stream>>>(
      kvh, kvl, (long)NK_ * C_, C_, wqh, wql, C_,
      qh, ql, nullptr, (long)NQ_ * C_, C_, C_, 0);
  // K = kv @ Wk^T
  gemm_bf16x3_kernel<<<dim3(C_ / 64, NK_ / 64, B_), 128, 0, stream>>>(
      kvh, kvl, (long)NK_ * C_, C_, wkh, wkl, C_,
      kh, kl, nullptr, (long)NK_ * C_, C_, C_, 0);
  // V^T = (kv @ Wv^T)^T  (stored [B, C, NK])
  gemm_bf16x3_kernel<<<dim3(C_ / 64, NK_ / 64, B_), 128, 0, stream>>>(
      kvh, kvl, (long)NK_ * C_, C_, wvh, wvl, C_,
      vth, vtl, nullptr, (long)C_ * NK_, NK_, C_, 1);

  // 3) attention
  attn_kernel<<<dim3(B_, NQ_ / 16), 256, 0, stream>>>(
      qh, ql, kh, kl, vth, vtl, mask, yh, yl);

  // 4) output projection -> fp32 d_out
  gemm_bf16x3_kernel<<<dim3(C_ / 64, NQ_ / 64, B_), 128, 0, stream>>>(
      yh, yl, (long)NQ_ * C_, C_, wph, wpl, C_,
      nullptr, nullptr, out, (long)NQ_ * C_, C_, C_, 2);
}
